// Attention_36704790512307
// MI455X (gfx1250) — compile-verified, matrix operand order checked by reading, not run on hardware
//
#include <hip/hip_runtime.h>
#include <hip/hip_bf16.h>

// Multi-head attention forward for MI455X (gfx1250, wave32, WMMA).
// B=2, S=4096, D_MODEL=512, H=8, D_K=64.
// All GEMMs (QKV projections, Q@K^T, P@V, row-sum-of-P, output projection)
// run on v_wmma_f32_16x16x32_f16 with fp32 accumulation; softmax is online
// (flash-attention style) in fp32 with ds_swizzle cross-lane max reduction.

#define BATCH 2
#define SEQ 4096
#define DMODEL 512
#define NHEAD 8
#define DK 64
#define MROWS (BATCH * SEQ)   // 8192

typedef __attribute__((ext_vector_type(16))) _Float16 v16h;
typedef __attribute__((ext_vector_type(8)))  float    v8f;

// ---------------------------------------------------------------------------
// WMMA wrapper: D = A(16x32 f16) * B(32x16 f16) + C(16x16 f32)
__device__ inline v8f wmma_f16(v16h a, v16h b, v8f c) {
  return __builtin_amdgcn_wmma_f32_16x16x32_f16(
      /*neg_a=*/false, a, /*neg_b=*/false, b,
      /*c_mod=*/(short)0, c, /*reuse_a=*/false, /*reuse_b=*/false);
}

// A/B fragment load from f16 operand stored row-major [rows, K].
// Lane layout (contiguous K order, used for A and B alike): r = lane%16 picks the row, lane/16 picks the
// 16-element K half; each lane loads 16 contiguous halves (32 bytes).
__device__ inline v16h frag_h(const _Float16* base, int ld, int row0, int k0,
                              int lane) {
  const int r = lane & 15, hh = lane >> 4;
  return *(const v16h*)(base + (size_t)(row0 + r) * ld + k0 + hh * 16);
}

// Same fragment pattern but source is f32: load 16 floats, convert to f16.
__device__ inline v16h frag_f32(const float* base, int ld, int row0, int k0,
                                int lane) {
  const int r = lane & 15, hh = lane >> 4;
  const float* p = base + (size_t)(row0 + r) * ld + k0 + hh * 16;
  v16h out;
#pragma unroll
  for (int i = 0; i < 16; ++i) out[i] = (_Float16)p[i];
  return out;
}

// Cross-lane XOR exchange within the wave via ds_swizzle (immediate pattern,
// group-of-32 mode: and=0x1f, or=0, xor=M). Masks < 16 stay inside each
// 16-lane half, which is exactly the row grouping of the WMMA C fragment.
template <int M>
__device__ inline float swz_xor(float x) {
  return __int_as_float(
      __builtin_amdgcn_ds_swizzle(__float_as_int(x), (M << 10) | 0x1f));
}
__device__ inline float redmax16(float x) {
  x = fmaxf(x, swz_xor<1>(x));
  x = fmaxf(x, swz_xor<2>(x));
  x = fmaxf(x, swz_xor<4>(x));
  x = fmaxf(x, swz_xor<8>(x));
  return x;
}

// ---------------------------------------------------------------------------
// f32 -> f16 weight conversion (scale folds 1/sqrt(d_k) into w_q).
__global__ void cvt_f16_kernel(const float* __restrict__ src,
                               _Float16* __restrict__ dst, int n, float scale) {
  int i = blockIdx.x * blockDim.x + threadIdx.x;
  const int stride = gridDim.x * blockDim.x;
  for (; i < n; i += stride) dst[i] = (_Float16)(src[i] * scale);
}

// ---------------------------------------------------------------------------
// Generic C = A[M=8192,K=512] * W[N=512,K=512]^T GEMM.
// Block = 256 threads = 8 waves arranged 4(m) x 2(n); block tile 128x64;
// each wave owns a 32x32 tile = 2x2 WMMA accumulators. Operand reuse across
// blocks is served by the 192MB L2 (whole A in f16 is 8MB).
// MODE 0: store f16 in per-head layout  Qp/Kp[((b*H+h)*S + s)*64 + d]
// MODE 1: store f16 transposed          Vt[((b*H+h)*64 + d)*S + s]
// MODE 2: A is f16, store f32 row-major [M, DMODEL] (final output)
template <int MODE>
__global__ __launch_bounds__(256) void gemm_wmma(const void* __restrict__ Av,
                                                 const _Float16* __restrict__ W,
                                                 void* __restrict__ dst) {
  const int lane = threadIdx.x & 31;
  const int wave = threadIdx.x >> 5;
  const int m0 = blockIdx.x * 128 + (wave >> 1) * 32;
  const int n0 = blockIdx.y * 64 + (wave & 1) * 32;

  v8f acc[2][2] = {};

  for (int kk = 0; kk < DMODEL; kk += 32) {
    v16h a0, a1;
    if constexpr (MODE == 2) {
      const _Float16* A = (const _Float16*)Av;
      a0 = frag_h(A, DMODEL, m0, kk, lane);
      a1 = frag_h(A, DMODEL, m0 + 16, kk, lane);
    } else {
      const float* A = (const float*)Av;
      a0 = frag_f32(A, DMODEL, m0, kk, lane);
      a1 = frag_f32(A, DMODEL, m0 + 16, kk, lane);
    }
    const v16h b0 = frag_h(W, DMODEL, n0, kk, lane);
    const v16h b1 = frag_h(W, DMODEL, n0 + 16, kk, lane);
    acc[0][0] = wmma_f16(a0, b0, acc[0][0]);
    acc[0][1] = wmma_f16(a0, b1, acc[0][1]);
    acc[1][0] = wmma_f16(a1, b0, acc[1][0]);
    acc[1][1] = wmma_f16(a1, b1, acc[1][1]);
  }

  // C layout: VGPR i, lane ℓ -> row = i + 8*(ℓ/16), col = ℓ%16.
#pragma unroll
  for (int mi = 0; mi < 2; ++mi)
#pragma unroll
    for (int ni = 0; ni < 2; ++ni)
#pragma unroll
      for (int i = 0; i < 8; ++i) {
        const int m = m0 + mi * 16 + i + ((lane >> 4) << 3);
        const int n = n0 + ni * 16 + (lane & 15);
        const float val = acc[mi][ni][i];
        if constexpr (MODE == 0) {
          const int b = m >> 12, s = m & (SEQ - 1);
          const int h = n >> 6, d = n & (DK - 1);
          ((_Float16*)dst)[(((size_t)(b * NHEAD + h) * SEQ + s) << 6) + d] =
              (_Float16)val;
        } else if constexpr (MODE == 1) {
          const int b = m >> 12, s = m & (SEQ - 1);
          const int h = n >> 6, d = n & (DK - 1);
          ((_Float16*)dst)[(((size_t)(b * NHEAD + h) * DK + d) << 12) + s] =
              (_Float16)val;
        } else {
          ((float*)dst)[(size_t)m * DMODEL + n] = val;
        }
      }
}

// ---------------------------------------------------------------------------
// Flash attention: one wave = 16 query rows of one (b,h) head.
// Streams keys in blocks of 64: 8 WMMA for the 16x64 score tile, online
// softmax (f32 running max via ds_swizzle reduction), LDS round-trip to turn
// the C-layout probability tile into A fragments, 2 WMMA against an all-ones
// B matrix for the row sums, 8 WMMA for P@V into the 16x64 f32 accumulator.
// Q arrives pre-scaled by 1/sqrt(d_k) (folded into w_q).
#define PPITCH 80  // halves per staged-P row: 160B keeps every row 32B-aligned

__global__ __launch_bounds__(256) void flash_attn(
    const _Float16* __restrict__ Qp, const _Float16* __restrict__ Kp,
    const _Float16* __restrict__ Vt, _Float16* __restrict__ X) {
  __shared__ _Float16 pbuf[8 * 16 * PPITCH];  // 20 KB

  const int lane = threadIdx.x & 31;
  const int wave = threadIdx.x >> 5;
  const int w = blockIdx.x * 8 + wave;      // global wave id
  const int bh = w >> 8;                    // / (SEQ/16)
  const int q0 = (w & 255) << 4;            // query tile start
  const int r = lane & 15, hh = lane >> 4;

  const _Float16* Q = Qp + (size_t)bh * SEQ * DK;
  const _Float16* K = Kp + (size_t)bh * SEQ * DK;
  const _Float16* V = Vt + (size_t)bh * DK * SEQ;  // [64 x SEQ] (transposed)
  _Float16* lds = pbuf + wave * 16 * PPITCH;

  // Q fragments, loaded once (16 rows x 64 d, split in two K=32 halves).
  const v16h qf0 = frag_h(Q, DK, q0, 0, lane);
  const v16h qf1 = frag_h(Q, DK, q0, 32, lane);

  // all-ones B fragment for the row-sum WMMA
  v16h ones;
#pragma unroll
  for (int i = 0; i < 16; ++i) ones[i] = (_Float16)1.0f;

  float mrow[8], lrow[8];
  v8f O[4] = {};
#pragma unroll
  for (int i = 0; i < 8; ++i) { mrow[i] = -1e30f; lrow[i] = 0.0f; }

  for (int j0 = 0; j0 < SEQ; j0 += 64) {
    // prefetch next key/value block; locality=3 -> WGP-scope prefetch, which
    // pulls the lines into all cache levels (this WGP consumes them next
    // iteration). Addresses past the last head still land inside the
    // workspace, and prefetches are side-effect free.
    __builtin_prefetch(K + (size_t)(j0 + 64 + lane) * DK, 0, 3);
    __builtin_prefetch(K + (size_t)(j0 + 96 + lane) * DK, 0, 3);
    __builtin_prefetch(V + (size_t)(lane * 2) * SEQ + j0 + 64, 0, 3);
    __builtin_prefetch(V + (size_t)(lane * 2 + 1) * SEQ + j0 + 64, 0, 3);

    // scores for 4 key tiles of 16 (keys j0 .. j0+63)
    v8f s[4];
#pragma unroll
    for (int c = 0; c < 4; ++c) {
      v8f z = {};
      s[c] = wmma_f16(qf0, frag_h(K, DK, j0 + c * 16, 0, lane), z);
      s[c] = wmma_f16(qf1, frag_h(K, DK, j0 + c * 16, 32, lane), s[c]);
    }

#pragma unroll
    for (int i = 0; i < 8; ++i) {
      // row max over all 64 keys of this block (16-lane cross reduction)
      float t = fmaxf(fmaxf(s[0][i], s[1][i]), fmaxf(s[2][i], s[3][i]));
      t = redmax16(t);
      const float mn = fmaxf(mrow[i], t);
      const float alpha = __expf(mrow[i] - mn);
      mrow[i] = mn;
      lrow[i] *= alpha;
#pragma unroll
      for (int t2 = 0; t2 < 4; ++t2) O[t2][i] *= alpha;  // rescale output acc
      // stage P (C layout -> row-major f16) in this wave's LDS region
      const int row = i + (hh << 3);
#pragma unroll
      for (int c = 0; c < 4; ++c)
        lds[row * PPITCH + r + c * 16] = (_Float16)__expf(s[c][i] - mn);
    }
    __syncthreads();  // uniform across block; orders DS store -> DS load
    const v16h pf0 = *(const v16h*)(lds + r * PPITCH + hh * 16);
    const v16h pf1 = *(const v16h*)(lds + r * PPITCH + 32 + hh * 16);
    __syncthreads();

    // row sums of the (f16-quantized) P tile via ones-matrix WMMA:
    // every column of rs holds sum_k P[row,k] for this 64-key block.
    v8f rs = {};
    rs = wmma_f16(pf0, ones, rs);
    rs = wmma_f16(pf1, ones, rs);
#pragma unroll
    for (int i = 0; i < 8; ++i) lrow[i] += rs[i];

    // O[16x64] += P[16x64] * V[64x64]; V^T rows give contiguous B fragments.
#pragma unroll
    for (int t = 0; t < 4; ++t) {
      O[t] = wmma_f16(pf0, frag_h(V, SEQ, t * 16, j0, lane), O[t]);
      O[t] = wmma_f16(pf1, frag_h(V, SEQ, t * 16, j0 + 32, lane), O[t]);
    }
  }

  // finalize: divide by softmax denominator, write f16 X in [B, S, DMODEL]
  const int b = bh >> 3, h = bh & 7;
#pragma unroll
  for (int i = 0; i < 8; ++i) {
    const float inv = 1.0f / lrow[i];
    const int srow = q0 + i + (hh << 3);
    const size_t base = ((size_t)b * SEQ + srow) * DMODEL + h * DK + r;
#pragma unroll
    for (int t = 0; t < 4; ++t)
      X[base + t * 16] = (_Float16)(O[t][i] * inv);
  }
}

// ---------------------------------------------------------------------------
extern "C" void kernel_launch(void* const* d_in, const int* in_sizes, int n_in,
                              void* d_out, int out_size, void* d_ws,
                              size_t ws_size, hipStream_t stream) {
  const float* q  = (const float*)d_in[0];
  const float* k  = (const float*)d_in[1];
  const float* v  = (const float*)d_in[2];
  const float* wq = (const float*)d_in[3];
  const float* wk = (const float*)d_in[4];
  const float* wv = (const float*)d_in[5];
  const float* wo = (const float*)d_in[6];

  // workspace layout (f16 halves): 4 weights + Qp + Kp + Vt + X  (~36 MB)
  _Float16* ws = (_Float16*)d_ws;
  const size_t WSZ = (size_t)DMODEL * DMODEL;       // 262144
  const size_t PSZ = (size_t)MROWS * DMODEL;        // 4194304
  _Float16* wqh = ws;
  _Float16* wkh = wqh + WSZ;
  _Float16* wvh = wkh + WSZ;
  _Float16* woh = wvh + WSZ;
  _Float16* Qp  = woh + WSZ;
  _Float16* Kp  = Qp + PSZ;
  _Float16* Vt  = Kp + PSZ;
  _Float16* X   = Vt + PSZ;

  // 1) weights -> f16 (fold 1/sqrt(d_k)=0.125 into w_q)
  cvt_f16_kernel<<<256, 256, 0, stream>>>(wq, wqh, (int)WSZ, 0.125f);
  cvt_f16_kernel<<<256, 256, 0, stream>>>(wk, wkh, (int)WSZ, 1.0f);
  cvt_f16_kernel<<<256, 256, 0, stream>>>(wv, wvh, (int)WSZ, 1.0f);
  cvt_f16_kernel<<<256, 256, 0, stream>>>(wo, woh, (int)WSZ, 1.0f);

  // 2) QKV projections (V stored transposed per head for the PV GEMM)
  dim3 g(MROWS / 128, DMODEL / 64), blk(256);
  gemm_wmma<0><<<g, blk, 0, stream>>>((const void*)q, wqh, (void*)Qp);
  gemm_wmma<0><<<g, blk, 0, stream>>>((const void*)k, wkh, (void*)Kp);
  gemm_wmma<1><<<g, blk, 0, stream>>>((const void*)v, wvh, (void*)Vt);

  // 3) flash attention: 16 heads * 256 query tiles = 4096 waves
  flash_attn<<<512, 256, 0, stream>>>(Qp, Kp, Vt, X);

  // 4) output projection -> f32 d_out [B*S, DMODEL]
  gemm_wmma<2><<<g, blk, 0, stream>>>((const void*)X, woh, d_out);
}
